// NeuroKernel_69956427318000
// MI455X (gfx1250) — matrix operand order repaired and checked by reading, not compiled, not run on hardware
//
#include <hip/hip_runtime.h>
#include <math.h>

typedef __attribute__((ext_vector_type(16))) _Float16 v16h;
typedef __attribute__((ext_vector_type(8)))  _Float16 v8h;
typedef __attribute__((ext_vector_type(8)))  float    v8f;
typedef __attribute__((ext_vector_type(4)))  int      v4i;

// int4 in explicit address spaces for the async global->LDS builtin
typedef __attribute__((address_space(1))) v4i gv4i;   // global memory
typedef __attribute__((address_space(3))) v4i sv4i;   // LDS

#define NX      768
#define PAIRS   ((NX * (NX + 1)) / 2)      // 295296, divisible by 16 and 128
#define H1      1024
#define H2      128
#define KSTEPS  (H1 / 32)                  // 32 K-slices for the MLP GEMM
#define NTILES  (H2 / 16)                  // 8 N-tiles of 16
#define GSTEPS  (NX / 32)                  // 24 K-slices for the Gram GEMM
#define GT      (NX / 16)                  // 48 tiles per output dim

#define STAGE_HALFS (NTILES * 32 * 16)     // 4096 halfs = 8 KB per K-slice

// workspace layout (bytes)
#define W2F16_BYTES (H2 * H1 * 2)          // 262144, 256-aligned
#define KT_BYTES    (NX * NX * 2)          // 1179648

// ---- CDNA5 async global->LDS path (guarded; sync-staging fallback) --------
#if __has_builtin(__builtin_amdgcn_global_load_async_to_lds_b128)
#define USE_ASYNC_LDS 1
#else
#define USE_ASYNC_LDS 0
#endif

#if USE_ASYNC_LDS
# if __has_builtin(__builtin_amdgcn_s_wait_asynccnt)
#  define WAIT_ASYNC(n) __builtin_amdgcn_s_wait_asynccnt(n)
# else
#  define WAIT_ASYNC(n) asm volatile("s_wait_asynccnt %0" :: "i"(n) : "memory")
# endif
#else
# define WAIT_ASYNC(n) ((void)0)
#endif

// sigmoid(z) = 0.5 * tanh(z/2) + 0.5  -> single v_tanh_f32 on gfx1250
__device__ __forceinline__ float fast_sigmoid(float z) {
#if __has_builtin(__builtin_amdgcn_tanhf)
    return fmaf(0.5f, __builtin_amdgcn_tanhf(0.5f * z), 0.5f);
#else
    return __builtin_amdgcn_rcpf(1.0f + __expf(-z));
#endif
}

// ---------------------------------------------------------------------------
// Kernel 1: convert W2 (f32, row-major [128][1024]) into f16, pre-swizzled in
// WMMA B-fragment order. Fragment (s = K-slice, t = N-tile): lane L holds
// column n = t*16 + (L%16), K elements k = s*32 + (L/16)*8 + e (e < 8),
// k = s*32 + 16 + (L/16)*8 + (e-8) (e >= 8).
// Linear index = (((s*8 + t)*32 + L)*16 + e).
// ---------------------------------------------------------------------------
__global__ void prep_w2_kernel(const float* __restrict__ W2,
                               _Float16* __restrict__ w2f) {
    int idx = blockIdx.x * blockDim.x + threadIdx.x;   // 0 .. 131071
    int e = idx & 15;
    int L = (idx >> 4) & 31;
    int t = (idx >> 9) & 7;
    int s = idx >> 12;
    int n = t * 16 + (L & 15);
    int k = s * 32 + (e >> 3) * 16 + (L >> 4) * 8 + (e & 7);
    w2f[idx] = (_Float16)W2[n * H1 + k];
}

// ---------------------------------------------------------------------------
// Kernel 2: zero the f16 K^T staging buffer (must be done every launch).
// ---------------------------------------------------------------------------
__global__ void zero_kt_kernel(uint4* __restrict__ p) {
    p[blockIdx.x * blockDim.x + threadIdx.x] = uint4{0u, 0u, 0u, 0u};
}

// ---------------------------------------------------------------------------
// Kernel 3: fused pair-MLP. One wave handles 16 pairs (one WMMA M-tile);
// 8 waves per block share W2 B-fragments staged through LDS (double
// buffered, async global->LDS copies on the CDNA5 path).
// ---------------------------------------------------------------------------
__global__ void __launch_bounds__(256)
mlp_pairs_kernel(const float* __restrict__ x,
                 const float* __restrict__ W1,
                 const float* __restrict__ b1,
                 const _Float16* __restrict__ w2f,
                 const float* __restrict__ b2,
                 const float* __restrict__ W3,
                 const float* __restrict__ b3,
                 _Float16* __restrict__ KT) {
    __shared__ _Float16 lbuf[2][STAGE_HALFS];          // 2 x 8 KB

    const int tid   = threadIdx.x;
    const int lane  = tid & 31;
    const int wave  = tid >> 5;
    const int tile  = blockIdx.x * 8 + wave;   // 0 .. 18455
    const int m     = lane & 15;               // A-fragment row within tile
    const int hi    = lane >> 4;               // half-wave id
    const int kbase = hi * 8;                  // A-fragment K sub-offset

    // stage a K-slice of pre-swizzled W2 into LDS: 256 threads x 32 B = 8 KB
    auto issue_stage = [&](int s, int buf) {
        const char* g = (const char*)(w2f + (size_t)s * STAGE_HALFS) + tid * 32;
        char*       l = (char*)&lbuf[buf][0] + tid * 32;
#if USE_ASYNC_LDS
        __builtin_amdgcn_global_load_async_to_lds_b128((gv4i*)g,        (sv4i*)l,        0, 0);
        __builtin_amdgcn_global_load_async_to_lds_b128((gv4i*)(g + 16), (sv4i*)(l + 16), 0, 0);
#else
        ((uint4*)l)[0] = ((const uint4*)g)[0];
        ((uint4*)l)[1] = ((const uint4*)g)[1];
#endif
    };

    // ---- invert triangular index: pair p -> (i, j), j >= i --------------
    int p = tile * 16 + m;
    const double tn = 2.0 * (double)NX + 1.0;
    int i = (int)((tn - sqrt(tn * tn - 8.0 * (double)p)) * 0.5);
    if (i < 0) i = 0;
    if (i > NX - 1) i = NX - 1;
    while (i + 1 <= NX - 1 && ((i + 1) * NX - ((i + 1) * i) / 2) <= p) ++i;
    while (i > 0 && (i * NX - (i * (i - 1)) / 2) > p) --i;
    const int j = i + (p - (i * NX - (i * (i - 1)) / 2));
    const float xi = x[i];
    const float xj = x[j];

    // prologue: stages 0 and 1 in flight
    issue_stage(0, 0);
    issue_stage(1, 1);

    // ---- layer-2 GEMM: 8 f32 accumulators (N = 128 = 8 x 16) ------------
    v8f c0 = {}, c1 = {}, c2 = {}, c3 = {}, c4 = {}, c5 = {}, c6 = {}, c7 = {};

    for (int s = 0; s < KSTEPS; ++s) {
        // layer 1: 16 sigmoid activations in A-fragment order (overlaps copy)
        v16h a;
        const int k0 = s * 32 + kbase;
#pragma unroll
        for (int e = 0; e < 8; ++e) {
            int ka = k0 + e;
            float za = fmaf(xi, W1[2 * ka], fmaf(xj, W1[2 * ka + 1], b1[ka]));
            a[e] = (_Float16)fast_sigmoid(za);
            int kb = ka + 16;
            float zb = fmaf(xi, W1[2 * kb], fmaf(xj, W1[2 * kb + 1], b1[kb]));
            a[e + 8] = (_Float16)fast_sigmoid(zb);
        }

        // stage s complete (stage s+1 may still be in flight)
        if (s == KSTEPS - 1) { WAIT_ASYNC(0); } else { WAIT_ASYNC(2); }
        __syncthreads();

        const _Float16* lb = &lbuf[s & 1][0];
        v16h b0f = *(const v16h*)&lb[(0 * 32 + lane) * 16];
        v16h b1f = *(const v16h*)&lb[(1 * 32 + lane) * 16];
        v16h b2f = *(const v16h*)&lb[(2 * 32 + lane) * 16];
        v16h b3f = *(const v16h*)&lb[(3 * 32 + lane) * 16];
        v16h b4f = *(const v16h*)&lb[(4 * 32 + lane) * 16];
        v16h b5f = *(const v16h*)&lb[(5 * 32 + lane) * 16];
        v16h b6f = *(const v16h*)&lb[(6 * 32 + lane) * 16];
        v16h b7f = *(const v16h*)&lb[(7 * 32 + lane) * 16];

        c0 = __builtin_amdgcn_wmma_f32_16x16x32_f16(false, a, false, b0f, (short)0, c0, false, false);
        c1 = __builtin_amdgcn_wmma_f32_16x16x32_f16(false, a, false, b1f, (short)0, c1, false, false);
        c2 = __builtin_amdgcn_wmma_f32_16x16x32_f16(false, a, false, b2f, (short)0, c2, false, false);
        c3 = __builtin_amdgcn_wmma_f32_16x16x32_f16(false, a, false, b3f, (short)0, c3, false, false);
        c4 = __builtin_amdgcn_wmma_f32_16x16x32_f16(false, a, false, b4f, (short)0, c4, false, false);
        c5 = __builtin_amdgcn_wmma_f32_16x16x32_f16(false, a, false, b5f, (short)0, c5, false, false);
        c6 = __builtin_amdgcn_wmma_f32_16x16x32_f16(false, a, false, b6f, (short)0, c6, false, false);
        c7 = __builtin_amdgcn_wmma_f32_16x16x32_f16(false, a, false, b7f, (short)0, c7, false, false);

        __syncthreads();                       // all waves done reading buf
        if (s + 2 < KSTEPS) issue_stage(s + 2, s & 1);
    }

    // ---- layer 3: bias + ReLU + dot(W3) + cross-lane reduction ----------
    // C layout: lane L, reg r -> (pair row M = r + 8*hi, N = t*16 + L%16)
    float acc[8];
#pragma unroll
    for (int r = 0; r < 8; ++r) acc[r] = 0.0f;

    v8f* cs[8] = { &c0, &c1, &c2, &c3, &c4, &c5, &c6, &c7 };
#pragma unroll
    for (int t = 0; t < 8; ++t) {
        int n = t * 16 + m;
        float bb = b2[n];
        float ww = W3[n];
        v8f cv = *cs[t];
#pragma unroll
        for (int r = 0; r < 8; ++r) {
            float v = cv[r] + bb;
            v = v > 0.0f ? v : 0.0f;
            acc[r] = fmaf(v, ww, acc[r]);
        }
    }
    // sum over the 16 lanes of each half-wave (xor masks keep hi fixed)
#pragma unroll
    for (int r = 0; r < 8; ++r) {
        acc[r] += __shfl_xor(acc[r], 1, 32);
        acc[r] += __shfl_xor(acc[r], 2, 32);
        acc[r] += __shfl_xor(acc[r], 4, 32);
        acc[r] += __shfl_xor(acc[r], 8, 32);
    }

    // gather (i, j) for the 8 pairs this half-wave owns (full-wave shfl)
    int io[8], jo[8];
#pragma unroll
    for (int r = 0; r < 8; ++r) {
        int mm = hi * 8 + r;
        io[r] = __shfl(i, mm, 32);
        jo[r] = __shfl(j, mm, 32);
    }
    const float bias3 = b3[0];
    if (m == 0) {
#pragma unroll
        for (int r = 0; r < 8; ++r) {
            KT[(size_t)jo[r] * NX + io[r]] = (_Float16)(acc[r] + bias3);
        }
    }
}

// ---------------------------------------------------------------------------
// Kernel 4: out = K^T K = KT * KT^T, f16 WMMA with f32 accumulate.
// One wave computes one 16x16 output tile. Both A and B fragments are
// contiguous row loads from KT (row-major).
// ---------------------------------------------------------------------------
__global__ void gram_kernel(const _Float16* __restrict__ KT,
                            float* __restrict__ out) {
    const int lane = threadIdx.x & 31;
    const int wave = threadIdx.x >> 5;
    const int tid  = blockIdx.x * 8 + wave;    // 0 .. 2303
    const int mt   = tid / GT;
    const int nt   = tid % GT;
    const int lm   = lane & 15;
    const int hi   = lane >> 4;

    const _Float16* Arow = KT + (size_t)(mt * 16 + lm) * NX;  // A = KT row-major
    const _Float16* Brow = KT + (size_t)(nt * 16 + lm) * NX;  // B[k][n] = KT[n][k]

    v8f c = {};
    for (int s = 0; s < GSTEPS; ++s) {
        v16h a, b;
        // A-fragment: lane holds row m, contiguous K run (L/16)*16
        a = *(const v16h*)&Arow[s * 32 + hi * 16];
        // B-fragment: lane holds col n, contiguous K run (L/16)*16
        b = *(const v16h*)&Brow[s * 32 + hi * 16];
        if (s + 1 < GSTEPS) {
            __builtin_prefetch(&Arow[(s + 1) * 32], 0, 1);
            __builtin_prefetch(&Brow[(s + 1) * 32], 0, 1);
        }
        c = __builtin_amdgcn_wmma_f32_16x16x32_f16(false, a, false, b, (short)0, c, false, false);
    }

    // C layout: lane L, reg r -> row = mt*16 + r + 8*hi, col = nt*16 + lm
#pragma unroll
    for (int r = 0; r < 8; ++r) {
        out[(size_t)(mt * 16 + r + hi * 8) * NX + nt * 16 + lm] = c[r];
    }
}

// ---------------------------------------------------------------------------
extern "C" void kernel_launch(void* const* d_in, const int* in_sizes, int n_in,
                              void* d_out, int out_size, void* d_ws, size_t ws_size,
                              hipStream_t stream) {
    const float* x  = (const float*)d_in[0];
    const float* W1 = (const float*)d_in[1];
    const float* b1 = (const float*)d_in[2];
    const float* W2 = (const float*)d_in[3];
    const float* b2 = (const float*)d_in[4];
    const float* W3 = (const float*)d_in[5];
    const float* b3 = (const float*)d_in[6];
    float* out = (float*)d_out;

    _Float16* w2f = (_Float16*)d_ws;
    _Float16* KT  = (_Float16*)((char*)d_ws + W2F16_BYTES);

    (void)in_sizes; (void)n_in; (void)out_size; (void)ws_size;

    // 1) pre-swizzle W2 into f16 B-fragment order: 128*1024 = 131072 elements
    prep_w2_kernel<<<512, 256, 0, stream>>>(W2, w2f);
    // 2) zero KT (1179648 B / 16 B = 73728 uint4)
    zero_kt_kernel<<<288, 256, 0, stream>>>((uint4*)KT);
    // 3) fused MLP over all pairs: 18456 wave-tiles / 8 waves per block
    mlp_pairs_kernel<<<PAIRS / 128, 256, 0, stream>>>(x, W1, b1, w2f, b2, W3, b3, KT);
    // 4) Gram matrix: 48*48 = 2304 tiles / 8 waves per block
    gram_kernel<<<288, 256, 0, stream>>>(KT, out);
}
